// CTC_47605417509223
// MI455X (gfx1250) — matrix operand order repaired and checked by reading, not compiled, not run on hardware
//
#include <hip/hip_runtime.h>
#include <hip/hip_bf16.h>

typedef __attribute__((ext_vector_type(16))) __bf16 v16bf;
typedef __attribute__((ext_vector_type(8)))  __bf16 v8bf;
typedef __attribute__((ext_vector_type(8)))  float  v8f;

#define Bn    16
#define Tn    1024
#define Dn    512
#define Vn    5000
#define Ln    128
#define Sn    257           // 2*L+1
#define NT_W  313           // ceil(5000/16)
#define NT_G  9             // ceil(144/16) label tiles (J padded to 144)
#define JPAD  144
#define KC_N  16            // 512 / 32
#define NEGV  (-1e30f)

// ---------------- workspace layout (bytes) ----------------
#define O_WHI   0ull
#define O_WLO   5126144ull
#define O_WGHI  10252288ull
#define O_WGLO  12611584ull
#define O_LSE   14970880ull
#define O_GLOG  15036416ull
#define O_LOSS  24473600ull

// ---------------------------------------------------------------------------
// Pack W (512 x 5000 f32 row-major) into bf16 hi/lo B-fragment layout.
// B frag (bf16 32x16): lane = h*16+n; lane holds K = h*16..h*16+15;
// VGPR j holds pair (K = h*16+2j, +1). Block (nt,kc) = 512 bf16 contiguous;
// element = lane*16 + j*2 + parity  -> per-lane 32B contiguous (b128 x2).
// ---------------------------------------------------------------------------
__global__ __launch_bounds__(256)
void packW_kernel(const float* __restrict__ W, __bf16* __restrict__ Whi,
                  __bf16* __restrict__ Wlo) {
  long id = (long)blockIdx.x * 256 + threadIdx.x;
  if (id >= (long)NT_W * 4096) return;
  int j    = (int)(id & 7);
  int lane = (int)((id >> 3) & 31);
  int kc   = (int)((id >> 8) & 15);
  int nt   = (int)(id >> 12);
  int n = lane & 15, h = lane >> 4;
  int k = kc * 32 + h * 16 + j * 2;
  int col = nt * 16 + n;
  float w0 = 0.f, w1 = 0.f;
  if (col < Vn) {
    w0 = W[(size_t)k * Vn + col];
    w1 = W[(size_t)(k + 1) * Vn + col];
  }
  size_t base = (((size_t)nt * 16 + kc) * 512) + (size_t)lane * 16 + j * 2;
  __bf16 h0 = (__bf16)w0, h1 = (__bf16)w1;
  Whi[base] = h0;                 Whi[base + 1] = h1;
  Wlo[base] = (__bf16)(w0 - (float)h0);
  Wlo[base + 1] = (__bf16)(w1 - (float)h1);
}

__global__ __launch_bounds__(256)
void packWg_kernel(const float* __restrict__ W, const int* __restrict__ ys_pad,
                   __bf16* __restrict__ Wghi, __bf16* __restrict__ Wglo) {
  long id = (long)blockIdx.x * 256 + threadIdx.x;
  if (id >= (long)Bn * NT_G * 4096) return;
  int j    = (int)(id & 7);
  int lane = (int)((id >> 3) & 31);
  int kc   = (int)((id >> 8) & 15);
  int ntb  = (int)(id >> 12);
  int nt = ntb % NT_G, b = ntb / NT_G;
  int n = lane & 15, h = lane >> 4;
  int k = kc * 32 + h * 16 + j * 2;
  int jj = nt * 16 + n;
  float w0 = 0.f, w1 = 0.f;
  if (jj <= 2 * Ln) {
    int col = (jj == 0) ? 0 : ys_pad[b * Ln + jj - 1];
    w0 = W[(size_t)k * Vn + col];
    w1 = W[(size_t)(k + 1) * Vn + col];
  }
  size_t base = (((size_t)(b * NT_G + nt) * 16 + kc) * 512) + (size_t)lane * 16 + j * 2;
  __bf16 h0 = (__bf16)w0, h1 = (__bf16)w1;
  Wghi[base] = h0;                 Wghi[base + 1] = h1;
  Wglo[base] = (__bf16)(w0 - (float)h0);
  Wglo[base + 1] = (__bf16)(w1 - (float)h1);
}

// ---------------------------------------------------------------------------
// Fused GEMM + online-logsumexp + label-logit gather.
// 32 rows (two 16-row tiles) per workgroup: each B fragment feeds 6 WMMAs.
// bf16-split: C = Ahi*Bhi + Ahi*Blo + Alo*Bhi, independent accumulators.
// A in LDS with XOR chunk swizzle (16B granularity) -> uniform 2-way banks.
// ---------------------------------------------------------------------------
#define SWZ(m, k) (((m) << 9) | (((((k) >> 3) ^ ((m) & 15)) << 3) | ((k) & 7)))

__global__ __launch_bounds__(256)
void gemm_lse_kernel(const float* __restrict__ hs, const float* __restrict__ bias,
                     const int* __restrict__ ys_pad,
                     const __bf16* __restrict__ Whi, const __bf16* __restrict__ Wlo,
                     const __bf16* __restrict__ Wghi, const __bf16* __restrict__ Wglo,
                     float* __restrict__ lse_out, float* __restrict__ glog) {
  __shared__ __attribute__((aligned(16))) char smem[65536];
  __bf16* Ahi = (__bf16*)smem;             // 32 rows x 512 bf16 = 32KB
  __bf16* Alo = (__bf16*)(smem + 32768);   // 32KB

  const int mt2  = blockIdx.x;             // 0..511, 32 rows each
  const int tid  = threadIdx.x;
  const int lane = tid & 31;
  const int wave = tid >> 5;
  const size_t rowbase = (size_t)mt2 * 32;
  const int bidx = mt2 >> 5;               // batch (32 blocks per batch)

  // Stage & split A (32 x 512 f32 -> bf16 hi/lo, XOR-swizzled), coalesced.
  for (int idx = tid; idx < 32 * Dn; idx += 256) {
    int m = idx >> 9, k = idx & 511;
    float x = hs[(rowbase + m) * Dn + k];
    __bf16 hi = (__bf16)x;
    int si = SWZ(m, k);
    Ahi[si] = hi;
    Alo[si] = (__bf16)(x - (float)hi);
  }
  __syncthreads();

  const int l15 = lane & 15;
  const int h   = lane >> 4;

  float m8a[8], s8a[8], m8b[8], s8b[8];
#pragma unroll
  for (int i = 0; i < 8; i++) {
    m8a[i] = -3.0e38f; s8a[i] = 0.f;
    m8b[i] = -3.0e38f; s8b[i] = 0.f;
  }

  for (int nt = wave; nt < NT_W + NT_G; nt += 8) {
    const __bf16 *bh, *bl;
    if (nt < NT_W) {
      bh = Whi + (size_t)nt * 16 * 512;
      bl = Wlo + (size_t)nt * 16 * 512;
    } else {
      size_t off = (size_t)(bidx * NT_G + (nt - NT_W)) * 16 * 512;
      bh = Wghi + off;
      bl = Wglo + off;
    }
    v8f c0h = {0.f,0.f,0.f,0.f,0.f,0.f,0.f,0.f};   // tile0: Ahi*Bhi
    v8f c0l = {0.f,0.f,0.f,0.f,0.f,0.f,0.f,0.f};   // tile0: Ahi*Blo
    v8f c0x = {0.f,0.f,0.f,0.f,0.f,0.f,0.f,0.f};   // tile0: Alo*Bhi
    v8f c1h = {0.f,0.f,0.f,0.f,0.f,0.f,0.f,0.f};   // tile1 terms
    v8f c1l = {0.f,0.f,0.f,0.f,0.f,0.f,0.f,0.f};
    v8f c1x = {0.f,0.f,0.f,0.f,0.f,0.f,0.f,0.f};
#pragma unroll 2
    for (int kc = 0; kc < KC_N; kc++) {
      // logical A chunks for this lane: c = kc*4 + 2h (grp0), kc*4 + 2h + 1 (grp1)
      int e0 = (((kc * 4 + 2 * h)     ^ l15) << 3);
      int e1 = (((kc * 4 + 2 * h + 1) ^ l15) << 3);
      int r0 = (l15) << 9;           // tile0 row base
      int r1 = (16 + l15) << 9;      // tile1 row base ((16+l15)&15 == l15)
      v8bf a0h_0 = *(const v8bf*)&Ahi[r0 + e0];
      v8bf a0h_1 = *(const v8bf*)&Ahi[r0 + e1];
      v8bf a0l_0 = *(const v8bf*)&Alo[r0 + e0];
      v8bf a0l_1 = *(const v8bf*)&Alo[r0 + e1];
      v8bf a1h_0 = *(const v8bf*)&Ahi[r1 + e0];
      v8bf a1h_1 = *(const v8bf*)&Ahi[r1 + e1];
      v8bf a1l_0 = *(const v8bf*)&Alo[r1 + e0];
      v8bf a1l_1 = *(const v8bf*)&Alo[r1 + e1];
      v16bf a0hi = __builtin_shufflevector(a0h_0, a0h_1, 0,1,2,3,4,5,6,7,8,9,10,11,12,13,14,15);
      v16bf a0lo = __builtin_shufflevector(a0l_0, a0l_1, 0,1,2,3,4,5,6,7,8,9,10,11,12,13,14,15);
      v16bf a1hi = __builtin_shufflevector(a1h_0, a1h_1, 0,1,2,3,4,5,6,7,8,9,10,11,12,13,14,15);
      v16bf a1lo = __builtin_shufflevector(a1l_0, a1l_1, 0,1,2,3,4,5,6,7,8,9,10,11,12,13,14,15);
      v16bf b_hi = *(const v16bf*)(bh + (size_t)kc * 512 + lane * 16);
      v16bf b_lo = *(const v16bf*)(bl + (size_t)kc * 512 + lane * 16);
      c0h = __builtin_amdgcn_wmma_f32_16x16x32_bf16(false, a0hi, false, b_hi, (short)0, c0h, false, false);
      c1h = __builtin_amdgcn_wmma_f32_16x16x32_bf16(false, a1hi, false, b_hi, (short)0, c1h, false, false);
      c0l = __builtin_amdgcn_wmma_f32_16x16x32_bf16(false, a0hi, false, b_lo, (short)0, c0l, false, false);
      c1l = __builtin_amdgcn_wmma_f32_16x16x32_bf16(false, a1hi, false, b_lo, (short)0, c1l, false, false);
      c0x = __builtin_amdgcn_wmma_f32_16x16x32_bf16(false, a0lo, false, b_hi, (short)0, c0x, false, false);
      c1x = __builtin_amdgcn_wmma_f32_16x16x32_bf16(false, a1lo, false, b_hi, (short)0, c1x, false, false);
    }
    if (nt < NT_W) {
      int col = nt * 16 + l15;
      if (col < Vn) {
        float bv = bias[col];
#pragma unroll
        for (int i = 0; i < 8; i++) {     // D: VGPR i -> row i+8h, col l15
          float x0 = (c0h[i] + c0l[i]) + c0x[i] + bv;
          float x1 = (c1h[i] + c1l[i]) + c1x[i] + bv;
          float nm0 = fmaxf(m8a[i], x0);
          s8a[i] = s8a[i] * __expf(m8a[i] - nm0) + __expf(x0 - nm0);
          m8a[i] = nm0;
          float nm1 = fmaxf(m8b[i], x1);
          s8b[i] = s8b[i] * __expf(m8b[i] - nm1) + __expf(x1 - nm1);
          m8b[i] = nm1;
        }
      }
    } else {
      int j = (nt - NT_W) * 16 + l15;     // 0..143
      float bv = 0.f;
      if (j <= 2 * Ln) {
        int col = (j == 0) ? 0 : ys_pad[bidx * Ln + j - 1];
        bv = bias[col];
      }
#pragma unroll
      for (int i = 0; i < 8; i++) {
        size_t row0 = rowbase + i + 8 * h;
        size_t row1 = rowbase + 16 + i + 8 * h;
        glog[row0 * JPAD + j] = (c0h[i] + c0l[i]) + c0x[i] + bv;
        glog[row1 * JPAD + j] = (c1h[i] + c1l[i]) + c1x[i] + bv;
      }
    }
  }

  // Butterfly-reduce (m,s) across the 16 lanes of each half (no LDS).
#pragma unroll
  for (int mask = 1; mask < 16; mask <<= 1) {
#pragma unroll
    for (int i = 0; i < 8; i++) {
      float mo, so, nm;
      mo = __shfl_xor(m8a[i], mask, 32);
      so = __shfl_xor(s8a[i], mask, 32);
      nm = fmaxf(m8a[i], mo);
      s8a[i] = s8a[i] * __expf(m8a[i] - nm) + so * __expf(mo - nm);
      m8a[i] = nm;
      mo = __shfl_xor(m8b[i], mask, 32);
      so = __shfl_xor(s8b[i], mask, 32);
      nm = fmaxf(m8b[i], mo);
      s8b[i] = s8b[i] * __expf(m8b[i] - nm) + so * __expf(mo - nm);
      m8b[i] = nm;
    }
  }

  // Alias reduction scratch over the (now dead) A staging buffer.
  __syncthreads();                         // all waves done reading A
  float* red_m = (float*)smem;             // [8 waves][32 rows]
  float* red_s = (float*)(smem + 4096);
  if (l15 == 0) {
#pragma unroll
    for (int i = 0; i < 8; i++) {
      red_m[wave * 32 + h * 8 + i]      = m8a[i];
      red_s[wave * 32 + h * 8 + i]      = s8a[i];
      red_m[wave * 32 + 16 + h * 8 + i] = m8b[i];
      red_s[wave * 32 + 16 + h * 8 + i] = s8b[i];
    }
  }
  __syncthreads();
  if (tid < 32) {
    float mm = -3.0e38f, ss = 0.f;
#pragma unroll
    for (int w = 0; w < 8; w++) {
      float mo = red_m[w * 32 + tid], so = red_s[w * 32 + tid];
      float nm = fmaxf(mm, mo);
      ss = ss * __expf(mm - nm) + so * __expf(mo - nm);
      mm = nm;
    }
    lse_out[rowbase + tid] = mm + __logf(ss);
  }
}

// ---------------------------------------------------------------------------
// CTC alpha recursion: one block per batch, S=257 states in LDS, T steps.
// ---------------------------------------------------------------------------
__global__ __launch_bounds__(256)
void ctc_alpha_kernel(const float* __restrict__ glog, const float* __restrict__ lse,
                      const int* __restrict__ ys_pad, const int* __restrict__ ys_lens,
                      const int* __restrict__ hlens, float* __restrict__ losses) {
  __shared__ float Abuf[2][Sn + 3];
  __shared__ unsigned char allow[Sn + 3];
  const int b = blockIdx.x, tid = threadIdx.x;

  for (int s = tid; s < Sn; s += 256) {
    bool al = false;
    if (s & 1) {
      int i = (s - 1) >> 1;
      al = (i == 0) || (ys_pad[b * Ln + i] != ys_pad[b * Ln + i - 1]);
    }
    allow[s] = al;
  }
  const float* g0 = glog + (size_t)(b * Tn) * JPAD;
  float l0 = lse[b * Tn];
  for (int s = tid; s < Sn; s += 256) {
    int j = (s & 1) ? ((s + 1) >> 1) : 0;
    Abuf[0][s] = (s <= 1) ? (g0[j] - l0) : NEGV;
  }
  __syncthreads();

  int hl = hlens[b]; if (hl > Tn) hl = Tn;   // keep = t < hlens
  int cur = 0;
  for (int t = 1; t < hl; t++) {
    int nxt = cur ^ 1;
    const float* gr = glog + (size_t)(b * Tn + t) * JPAD;
    float ls = lse[b * Tn + t];
    for (int s = tid; s < Sn; s += 256) {
      float a1 = Abuf[cur][s];
      float a2 = (s >= 1) ? Abuf[cur][s - 1] : NEGV;
      float a3 = (s >= 2 && allow[s]) ? Abuf[cur][s - 2] : NEGV;
      float m = fmaxf(fmaxf(a1, a2), a3);
      m = fmaxf(m, NEGV);
      float v = m + __logf(__expf(a1 - m) + __expf(a2 - m) + __expf(a3 - m));
      int j = (s & 1) ? ((s + 1) >> 1) : 0;
      Abuf[nxt][s] = v + (gr[j] - ls);
    }
    __syncthreads();
    cur = nxt;
  }
  if (tid == 0) {
    int yl = ys_lens[b];
    int i1 = 2 * yl;
    int i2 = (i1 > 0) ? i1 - 1 : 0;
    float aa = Abuf[cur][i1], bb = Abuf[cur][i2];
    float m = fmaxf(aa, bb);
    float ll = m + __logf(__expf(aa - m) + __expf(bb - m));
    float loss = -ll;
    bool keep = (loss == loss) && (fabsf(loss) <= 3.0e38f) && (loss < 1e29f);
    losses[b] = keep ? loss : 0.f;
  }
}

__global__ void final_kernel(const float* __restrict__ losses,
                             const int* __restrict__ ys_lens, float* __restrict__ out) {
  if (threadIdx.x == 0 && blockIdx.x == 0) {
    float sl = 0.f; long syl = 0;
    for (int i = 0; i < Bn; i++) { sl += losses[i]; syl += ys_lens[i]; }
    out[0] = sl / (float)syl;
  }
}

extern "C" void kernel_launch(void* const* d_in, const int* in_sizes, int n_in,
                              void* d_out, int out_size, void* d_ws, size_t ws_size,
                              hipStream_t stream) {
  (void)in_sizes; (void)n_in; (void)out_size; (void)ws_size;
  const float* hs      = (const float*)d_in[0];
  const int*   hlens   = (const int*)d_in[1];
  const int*   ys_pad  = (const int*)d_in[2];
  const int*   ys_lens = (const int*)d_in[3];
  const float* W       = (const float*)d_in[4];
  const float* bias    = (const float*)d_in[5];
  float* out = (float*)d_out;
  char*  ws  = (char*)d_ws;                 // needs ~23.4 MB

  __bf16* Whi  = (__bf16*)(ws + O_WHI);
  __bf16* Wlo  = (__bf16*)(ws + O_WLO);
  __bf16* Wghi = (__bf16*)(ws + O_WGHI);
  __bf16* Wglo = (__bf16*)(ws + O_WGLO);
  float*  lse  = (float*)(ws + O_LSE);
  float*  glog = (float*)(ws + O_GLOG);
  float*  loss = (float*)(ws + O_LOSS);

  packW_kernel <<<(NT_W * 4096) / 256, 256, 0, stream>>>(W, Whi, Wlo);
  packWg_kernel<<<(Bn * NT_G * 4096) / 256, 256, 0, stream>>>(W, ys_pad, Wghi, Wglo);
  gemm_lse_kernel<<<(Bn * Tn) / 32, 256, 0, stream>>>(hs, bias, ys_pad, Whi, Wlo,
                                                      Wghi, Wglo, lse, glog);
  ctc_alpha_kernel<<<Bn, 256, 0, stream>>>(glog, lse, ys_pad, ys_lens, hlens, loss);
  final_kernel<<<1, 64, 0, stream>>>(loss, ys_lens, out);
}
